// MPN_77352361001089
// MI455X (gfx1250) — matrix operand order repaired and checked by reading, not compiled, not run on hardware
//
#include <hip/hip_runtime.h>
#include <hip/hip_bf16.h>

typedef _Float16 v16h __attribute__((ext_vector_type(16)));
typedef _Float16 v8h  __attribute__((ext_vector_type(8)));
typedef _Float16 v4h  __attribute__((ext_vector_type(4)));
typedef float    v8f  __attribute__((ext_vector_type(8)));

#define WMMA_F16(a, b, c) \
  __builtin_amdgcn_wmma_f32_16x16x32_f16(false, (a), false, (b), (short)0, (c), false, false)

// Wave-private LDS ordering: DS ops from one wave complete in order; the asm
// memory clobber stops the compiler reordering, s_wait_dscnt 0 drains DS.
__device__ __forceinline__ void wave_lds_fence() {
  asm volatile("s_wait_dscnt 0" ::: "memory");
}

// A fragment (16x32 f16, M=rows of 16 edges): lane<16 holds row m=lane,
// K kb+0..15; lane>=16 holds row m=lane-16, K kb+16..31.
__device__ __forceinline__ v16h load_a_frag(const _Float16* base, int stride,
                                            int kbase, int lane) {
  const int m   = lane & 15;
  const int off = (lane >> 4) << 4;
  const _Float16* p = base + m * stride + kbase + off;
  v8h lo = *((const v8h*)p);
  v8h hi = *((const v8h*)(p + 8));
  return __builtin_shufflevector(lo, hi, 0, 1, 2, 3, 4, 5, 6, 7,
                                 8, 9, 10, 11, 12, 13, 14, 15);
}

// B fragment (32x16 f16): tile stored [n][k] (n=0..15, k=0..31).
// lane<16: column n=lane, K 0..15; lane>=16: column n=lane-16, K 16..31.
__device__ __forceinline__ v16h load_b_frag(const _Float16* tile, int lane) {
  const int n  = lane & 15;
  const int kc = (lane >> 4) << 4;
  return *((const v16h*)(tile + n * 32 + kc));
}

__global__ void zero_f32_kernel(float* __restrict__ p, long long n) {
  long long i = (long long)blockIdx.x * blockDim.x + threadIdx.x;
  if (i < n) p[i] = 0.0f;
}

__global__ __launch_bounds__(256) void mpn_wmma_kernel(
    const float* __restrict__ x_node, const float* __restrict__ x_edge,
    const int* __restrict__ src, const int* __restrict__ dst,
    const float* __restrict__ We1, const float* __restrict__ be1,
    const float* __restrict__ We2, const float* __restrict__ be2,
    const float* __restrict__ Wn1, const float* __restrict__ bn1,
    const float* __restrict__ Wn2, const float* __restrict__ bn2,
    float* __restrict__ out_nm, float* __restrict__ out_em, long long E) {
  // Weight B-tiles, f16, [tile][n][k].
  __shared__ __align__(16) _Float16 w1[6 * 512];  // GEMM1: 2 N-tiles x 3 K-chunks (70->96 pad)
  __shared__ __align__(16) _Float16 w2[1 * 512];  // GEMM2: N 6->16 pad, K=32
  __shared__ __align__(16) _Float16 w3[8 * 512];  // GEMM3: 4 N-tiles x 2 K-chunks (38->64 pad)
  __shared__ __align__(16) _Float16 w4[4 * 512];  // GEMM4: 2 N-tiles x 2 K-chunks
  // Per-wave activation slabs.
  __shared__ __align__(16) _Float16 slabA[8 * 16 * 96];  // e_in / h2 reuse
  __shared__ __align__(16) _Float16 slabB[8 * 16 * 64];  // n_in

  const int tid  = threadIdx.x;
  const int lane = tid & 31;
  const int wave = tid >> 5;

  // ---- stage weights (once per block) ----
  for (int i = tid; i < 6 * 512; i += 256) {
    int k = i & 31, n = (i >> 5) & 15, tc = i >> 9;  // tc = t*3 + c
    int t = tc / 3, c = tc % 3;
    int kk = c * 32 + k;
    w1[i] = (kk < 70) ? (_Float16)We1[kk * 32 + t * 16 + n] : (_Float16)0.0f;
  }
  for (int i = tid; i < 512; i += 256) {
    int k = i & 31, n = i >> 5;
    w2[i] = (n < 6) ? (_Float16)We2[k * 6 + n] : (_Float16)0.0f;
  }
  for (int i = tid; i < 8 * 512; i += 256) {
    int k = i & 31, n = (i >> 5) & 15, tc = i >> 9;  // tc = t*2 + c
    int t = tc >> 1, c = tc & 1;
    int kk = c * 32 + k;
    w3[i] = (kk < 38) ? (_Float16)Wn1[kk * 64 + t * 16 + n] : (_Float16)0.0f;
  }
  for (int i = tid; i < 4 * 512; i += 256) {
    int k = i & 31, n = (i >> 5) & 15, tc = i >> 9;
    int t = tc >> 1, c = tc & 1;
    w4[i] = (_Float16)Wn2[(c * 32 + k) * 32 + t * 16 + n];
  }
  __syncthreads();

  _Float16* eA = slabA + wave * (16 * 96);
  _Float16* eB = slabB + wave * (16 * 64);
  const long long edgeBase = ((long long)blockIdx.x * 8 + wave) * 16;
  const int n0 = lane & 15;        // C/D: feature within N-tile
  const int mh = (lane >> 4) * 8;  // C/D: row offset (VGPR j -> row mh+j)

  // ---- build e_in[16x96] = [dst_x(32) | src_x(32) | x_edge(6) | 0(26)] ----
  {
    const int m = lane >> 1, h = lane & 1;
    long long e  = edgeBase + m;
    long long ec = (e < E) ? e : 0;
    const float4* dx = (const float4*)(x_node + (long long)dst[ec] * 32 + h * 16);
    const float4* sx = (const float4*)(x_node + (long long)src[ec] * 32 + h * 16);
#pragma unroll
    for (int i = 0; i < 4; ++i) {
      float4 v = dx[i];
      v4h hv = {(_Float16)v.x, (_Float16)v.y, (_Float16)v.z, (_Float16)v.w};
      *(v4h*)(eA + m * 96 + h * 16 + 4 * i) = hv;
      float4 s = sx[i];
      v4h hs = {(_Float16)s.x, (_Float16)s.y, (_Float16)s.z, (_Float16)s.w};
      *(v4h*)(eA + m * 96 + 32 + h * 16 + 4 * i) = hs;
    }
    const float* ex = x_edge + ec * 6 + h * 3;
#pragma unroll
    for (int i = 0; i < 3; ++i) eA[m * 96 + 64 + h * 3 + i] = (_Float16)ex[i];
#pragma unroll
    for (int i = 0; i < 13; ++i) eA[m * 96 + 70 + h * 13 + i] = (_Float16)0.0f;
  }
  wave_lds_fence();

  // ---- GEMM1: e_in[16x96] @ We1pad[96x32] ----
  v8f c1[2] = {};
#pragma unroll
  for (int c = 0; c < 3; ++c) {
    v16h a = load_a_frag(eA, 96, c * 32, lane);
#pragma unroll
    for (int t = 0; t < 2; ++t) {
      v16h b = load_b_frag(w1 + (t * 3 + c) * 512, lane);
      c1[t] = WMMA_F16(a, b, c1[t]);
    }
  }
  wave_lds_fence();
  // bias + ReLU -> h1 f16 into eA cols 0..31 (A layout for GEMM2)
#pragma unroll
  for (int t = 0; t < 2; ++t) {
    float bias = be1[t * 16 + n0];
#pragma unroll
    for (int j = 0; j < 8; ++j) {
      float v = c1[t][j] + bias;
      v = v > 0.0f ? v : 0.0f;
      eA[(mh + j) * 96 + t * 16 + n0] = (_Float16)v;
    }
  }
  wave_lds_fence();

  // ---- GEMM2: h1[16x32] @ We2pad[32x16] -> em ----
  v8f c2 = {};
  {
    v16h a = load_a_frag(eA, 96, 0, lane);
    v16h b = load_b_frag(w2, lane);
    c2 = WMMA_F16(a, b, c2);
  }
  // em: store fp32 output + stage f16 into n_in cols 32..37
  {
    float bias = (n0 < 6) ? be2[n0] : 0.0f;
#pragma unroll
    for (int j = 0; j < 8; ++j) {
      float v = c2[j] + bias;
      if (n0 < 6) {
        long long e = edgeBase + mh + j;
        if (e < E) out_em[e * 6 + n0] = v;
        eB[(mh + j) * 64 + 32 + n0] = (_Float16)v;
      }
    }
  }
  // n_in cols 0..31 = dst_x, cols 38..63 = 0
  {
    const int m = lane >> 1, h = lane & 1;
    long long e  = edgeBase + m;
    long long ec = (e < E) ? e : 0;
    const float4* dx = (const float4*)(x_node + (long long)dst[ec] * 32 + h * 16);
#pragma unroll
    for (int i = 0; i < 4; ++i) {
      float4 v = dx[i];
      v4h hv = {(_Float16)v.x, (_Float16)v.y, (_Float16)v.z, (_Float16)v.w};
      *(v4h*)(eB + m * 64 + h * 16 + 4 * i) = hv;
    }
#pragma unroll
    for (int i = 0; i < 13; ++i) eB[m * 64 + 38 + h * 13 + i] = (_Float16)0.0f;
  }
  wave_lds_fence();

  // ---- GEMM3: n_in[16x64] @ Wn1pad[64x64] ----
  v8f c3[4] = {};
#pragma unroll
  for (int c = 0; c < 2; ++c) {
    v16h a = load_a_frag(eB, 64, c * 32, lane);
#pragma unroll
    for (int t = 0; t < 4; ++t) {
      v16h b = load_b_frag(w3 + (t * 2 + c) * 512, lane);
      c3[t] = WMMA_F16(a, b, c3[t]);
    }
  }
  wave_lds_fence();
  // bias + ReLU -> h2 f16 into eA cols 0..63 (A layout for GEMM4)
#pragma unroll
  for (int t = 0; t < 4; ++t) {
    float bias = bn1[t * 16 + n0];
#pragma unroll
    for (int j = 0; j < 8; ++j) {
      float v = c3[t][j] + bias;
      v = v > 0.0f ? v : 0.0f;
      eA[(mh + j) * 96 + t * 16 + n0] = (_Float16)v;
    }
  }
  wave_lds_fence();

  // ---- GEMM4: h2[16x64] @ Wn2[64x32] -> msg ----
  v8f c4[2] = {};
#pragma unroll
  for (int c = 0; c < 2; ++c) {
    v16h a = load_a_frag(eA, 96, c * 32, lane);
#pragma unroll
    for (int t = 0; t < 2; ++t) {
      v16h b = load_b_frag(w4 + (t * 2 + c) * 512, lane);
      c4[t] = WMMA_F16(a, b, c4[t]);
    }
  }

  // ---- scatter-add msg onto dst nodes ----
#pragma unroll
  for (int j = 0; j < 8; ++j) {
    long long e = edgeBase + mh + j;
    if (e < E) {
      long long d = dst[e];
#pragma unroll
      for (int t = 0; t < 2; ++t) {
        float v = c4[t][j] + bn2[t * 16 + n0];
        atomicAdd(out_nm + d * 32 + t * 16 + n0, v);
      }
    }
  }
}

extern "C" void kernel_launch(void* const* d_in, const int* in_sizes, int n_in,
                              void* d_out, int out_size, void* d_ws, size_t ws_size,
                              hipStream_t stream) {
  const float* x_node = (const float*)d_in[0];
  const float* x_edge = (const float*)d_in[1];
  const int*   src    = (const int*)d_in[2];
  const int*   dst    = (const int*)d_in[3];
  const float* We1    = (const float*)d_in[4];
  const float* be1    = (const float*)d_in[5];
  const float* We2    = (const float*)d_in[6];
  const float* be2    = (const float*)d_in[7];
  const float* Wn1    = (const float*)d_in[8];
  const float* bn1    = (const float*)d_in[9];
  const float* Wn2    = (const float*)d_in[10];
  const float* bn2    = (const float*)d_in[11];

  const long long E  = in_sizes[3];          // number of edges
  const long long Nn = in_sizes[0] / 32;     // number of nodes
  float* out_nm = (float*)d_out;             // [Nn,32]
  float* out_em = (float*)d_out + Nn * 32;   // [E,6]

  const long long nmElems = Nn * 32;
  zero_f32_kernel<<<(int)((nmElems + 255) / 256), 256, 0, stream>>>(out_nm, nmElems);

  const int blocks = (int)((E + 127) / 128);  // 128 edges per 256-thread block
  mpn_wmma_kernel<<<blocks, 256, 0, stream>>>(x_node, x_edge, src, dst,
                                              We1, be1, We2, be2,
                                              Wn1, bn1, Wn2, bn2,
                                              out_nm, out_em, E);
}
